// DCNv4Dyhead_56959856279793
// MI455X (gfx1250) — matrix operand order repaired and checked by reading, not compiled, not run on hardware
//
#include <hip/hip_runtime.h>
#include <hip/hip_bf16.h>

typedef __attribute__((ext_vector_type(16))) _Float16 v16h;
typedef __attribute__((ext_vector_type(8)))  _Float16 v8h;
typedef __attribute__((ext_vector_type(8)))  float    v8f;
typedef __attribute__((ext_vector_type(4)))  int      v4i;

#if defined(__AMDGCN__)
typedef __attribute__((address_space(1))) v4i gv4i_t;
typedef __attribute__((address_space(3))) v4i lv4i_t;
#endif

#define CC   256
#define NG   4
#define KK   9
#define OMC  108   // G*K*3
#define OMP  112   // padded to 7*16
#define NB   2
#define PTOT 5376  // 4096+1024+256 pixels per batch

// ---------------------------------------------------------------------------
// prep: convert w_om -> f16 [112][2304] with K = kpos*256 + c ; w_proj -> f16
// ---------------------------------------------------------------------------
__global__ void k_prep(const float* __restrict__ w_om, const float* __restrict__ w_proj,
                       _Float16* __restrict__ w_om16, _Float16* __restrict__ w_proj16) {
    int i = blockIdx.x * 256 + threadIdx.x;
    const int total1 = OMP * 2304;
    if (i < total1) {
        int oc = i / 2304, k = i % 2304;
        int kpos = k >> 8, c = k & 255;
        float v = (oc < OMC) ? w_om[((size_t)oc * 256 + c) * 9 + kpos] : 0.0f;
        w_om16[i] = (_Float16)v;
    }
    int j = i - total1;
    if (j >= 0 && j < 256 * 256) w_proj16[j] = (_Float16)w_proj[j];
}

// ---------------------------------------------------------------------------
// pooled spatial mean per (b,c): works on [b][C][planeStride] layouts
// ---------------------------------------------------------------------------
__global__ void k_pool(const float* __restrict__ in, float* __restrict__ out,
                       int planeStride, int baseOff, int HW) {
    int c = blockIdx.x, b = blockIdx.y;
    const float* p = in + (size_t)(b * CC + c) * planeStride + baseOff;
    float s = 0.0f;
    for (int i = threadIdx.x; i < HW; i += 256) s += p[i];
    __shared__ float red[256];
    red[threadIdx.x] = s;
    __syncthreads();
    for (int st = 128; st > 0; st >>= 1) {
        if (threadIdx.x < st) red[threadIdx.x] += red[threadIdx.x + st];
        __syncthreads();
    }
    if (threadIdx.x == 0) out[b * CC + c] = red[0] / (float)HW;
}

// ---------------------------------------------------------------------------
// scale_att vector: s = clip(0.5*(W*mean + bias + 1), 0, 1)
// ---------------------------------------------------------------------------
__global__ void k_scalevec(const float* __restrict__ w, const float* __restrict__ bias,
                           const float* __restrict__ mean, float* __restrict__ s) {
    int o = threadIdx.x, b = blockIdx.x;
    float acc = bias[o];
    const float* wr = w + (size_t)o * 256;
    const float* mr = mean + b * 256;
    for (int c = 0; c < 256; ++c) acc += wr[c] * mr[c];
    acc = (acc + 1.0f) * 0.5f;
    s[b * CC + o] = fminf(fmaxf(acc, 0.0f), 1.0f);
}

// relu matvec for task_att branches
__global__ void k_matvec_relu(const float* __restrict__ w, const float* __restrict__ bias,
                              const float* __restrict__ mean, float* __restrict__ out) {
    int o = threadIdx.x, b = blockIdx.x;
    float acc = bias[o];
    const float* wr = w + (size_t)o * 256;
    const float* mr = mean + b * 256;
    for (int c = 0; c < 256; ++c) acc += wr[c] * mr[c];
    out[b * CC + o] = fmaxf(acc, 0.0f);
}

// ---------------------------------------------------------------------------
// apply per-(b,c) scale: f1 (B,C,HW) -> xs [b][C][PTOT] (f32, channel-major)
// plus xs16 [b][PTOT][C] (f16, pixel-major) for async LDS staging in conv
// ---------------------------------------------------------------------------
__global__ void k_applyscale(const float* __restrict__ x, const float* __restrict__ s,
                             float* __restrict__ xs, _Float16* __restrict__ xs16,
                             int HW, int baseOff, int P) {
    int idx = blockIdx.x * 256 + threadIdx.x;
    int total = NB * CC * HW;
    if (idx >= total) return;
    int p = idx % HW, bc = idx / HW;
    int b = bc >> 8, c = bc & 255;
    float v = x[idx] * s[bc];
    xs[(size_t)bc * P + baseOff + p] = v;
    xs16[((size_t)b * P + baseOff + p) * 256 + c] = (_Float16)v;
}

// ---------------------------------------------------------------------------
// conv3x3 (C=256 -> 108) as implicit GEMM via v_wmma_f32_16x16x32_f16
//  D(16 outch x 16 pix) = A(w_om16 16x2304) * B(im2col patch 2304x16)
//  block = 128 thr = 4 waves; wave w does M-tiles {w, w+4 (<7)}
//  patch staged via global_load_async_to_lds_b128 (512B contiguous per row)
// ---------------------------------------------------------------------------
__global__ void __launch_bounds__(128) k_conv3x3(
        const _Float16* __restrict__ xs16, const _Float16* __restrict__ w_om16,
        const float* __restrict__ b_om, float* __restrict__ om,
        int H, int W, int baseOff, int P) {
    __shared__ _Float16 patch[54 * 256];   // [row = dy*18+dx][c] f16, 27KB
    int b = blockIdx.y;
    int tilesPerRow = W >> 4;
    int h = blockIdx.x / tilesPerRow;
    int wbase = (blockIdx.x % tilesPerRow) << 4;
    int tid = threadIdx.x;
    int wave = tid >> 5, lane = tid & 31;
    int hi = lane >> 4, lo = lane & 15;

    // stage 54 pixel-rows (256 ch f16 each = 512B): one wave-wide async copy
    // per in-bounds row, zero-fill for padding rows (branch is wave-uniform)
    for (int r = wave; r < 54; r += 4) {
        int dy = r / 18, dx = r % 18;
        int y = h + dy - 1, x = wbase + dx - 1;
        _Float16* lp = &patch[r * 256 + lane * 8];
        if (y >= 0 && y < H && x >= 0 && x < W) {
            const _Float16* gp =
                xs16 + ((size_t)b * P + baseOff + y * W + x) * 256 + lane * 8;
#if defined(__gfx1250__) && __has_builtin(__builtin_amdgcn_global_load_async_to_lds_b128)
            __builtin_amdgcn_global_load_async_to_lds_b128(
                (gv4i_t*)(void*)gp, (lv4i_t*)(void*)lp, 0, 0);
#else
            *(v8h*)lp = *(const v8h*)gp;
#endif
        } else {
            v8h z = {};
            *(v8h*)lp = z;
        }
    }
#if defined(__gfx1250__) && __has_builtin(__builtin_amdgcn_global_load_async_to_lds_b128)
#if __has_builtin(__builtin_amdgcn_s_wait_asynccnt)
    __builtin_amdgcn_s_wait_asynccnt(0);
#else
    asm volatile("s_wait_asynccnt 0" ::: "memory");
#endif
#endif
    __syncthreads();

    for (int mi = 0; mi < 2; ++mi) {
        int mt = wave + mi * 4;
        if (mt > 6) break;                 // wave-uniform
        v8f acc = {};
        for (int t = 0; t < 72; ++t) {     // K = 9 taps * 256 ch, 32 per step
            int kpos  = t >> 3;
            int cbase = (t & 7) << 5;
            v16h a, bf;
            // A frag: row = mt*16+lo ; K chunks {hi*8 .. +7} and {hi*8+16 ..}
            const _Float16* pa = w_om16 + (size_t)(mt * 16 + lo) * 2304 + t * 32 + hi * 8;
            ((v8h*)&a)[0] = *(const v8h*)pa;
            ((v8h*)&a)[1] = *(const v8h*)(pa + 16);
            // B frag: col = pixel lo ; K chunks {hi*8 .. +7} and {hi*8+16 ..}
            int pdy = kpos / 3;
            int pdx = lo + (kpos % 3);
            const _Float16* pb = &patch[(pdy * 18 + pdx) * 256 + cbase + hi * 8];
            ((v8h*)&bf)[0] = *(const v8h*)pb;
            ((v8h*)&bf)[1] = *(const v8h*)(pb + 16);
            acc = __builtin_amdgcn_wmma_f32_16x16x32_f16(false, a, false, bf,
                                                         (short)0, acc, false, false);
        }
        // D: vgpr r -> row hi*8+r, col lo ; only mt==6 has padded rows
        size_t obase = (size_t)b * OMC * P + baseOff + h * W + wbase + lo;
        if (mt < 6) {
            for (int r = 0; r < 8; ++r) {
                int oc = mt * 16 + hi * 8 + r;
                om[obase + (size_t)oc * P] = acc[r] + b_om[oc];
            }
        } else {
            for (int r = 0; r < 8; ++r) {
                int oc = 96 + hi * 8 + r;
                if (oc < OMC)
                    om[obase + (size_t)oc * P] = acc[r] + b_om[oc];
            }
        }
    }
}

// ---------------------------------------------------------------------------
// deformable sampling: one block per pixel, 256 thr = channels
// writes pixel-major f16 dcnT[b][p][c] for the proj GEMM
// ---------------------------------------------------------------------------
__global__ void __launch_bounds__(256) k_dcn(
        const float* __restrict__ xs, const float* __restrict__ om,
        _Float16* __restrict__ dcnT, int H, int W, int baseOff, int P) {
    __shared__ float oms[OMC];
    int p = blockIdx.x, b = blockIdx.y;
    int tid = threadIdx.x;
    if (tid < OMC) oms[tid] = om[(size_t)(b * OMC + tid) * P + baseOff + p];
    __syncthreads();
    int c = tid, g = c >> 6;
    int h = p / W, w = p % W;
    const float* xc = xs + (size_t)(b * CC + c) * P + baseOff;
    float acc = 0.0f;
#pragma unroll
    for (int k = 0; k < KK; ++k) {
        float oy = oms[g * 27 + k * 3 + 0];
        float ox = oms[g * 27 + k * 3 + 1];
        float m  = oms[g * 27 + k * 3 + 2];
        float py = (float)(h + k / 3 - 1) + oy;
        float px = (float)(w + k % 3 - 1) + ox;
        float y0 = floorf(py), x0 = floorf(px);
        float fy = py - y0, fx = px - x0;
        int y0i = (int)y0, x0i = (int)x0;
#pragma unroll
        for (int dy = 0; dy < 2; ++dy)
#pragma unroll
            for (int dx = 0; dx < 2; ++dx) {
                int yy = y0i + dy, xx = x0i + dx;
                if (yy >= 0 && yy < H && xx >= 0 && xx < W) {
                    float wt = (dy ? fy : 1.0f - fy) * (dx ? fx : 1.0f - fx) * m;
                    acc += wt * xc[yy * W + xx];
                }
            }
    }
    dcnT[((size_t)b * P + baseOff + p) * 256 + c] = (_Float16)acc;
}

// ---------------------------------------------------------------------------
// proj 1x1 GEMM (WMMA) fused with LayerNorm + sigmoid gating
// block = 256 thr = 8 waves; 16 pixels x 256 out-channels per block
// ---------------------------------------------------------------------------
__global__ void __launch_bounds__(256) k_projlngate(
        const _Float16* __restrict__ dcnT, const _Float16* __restrict__ w_proj16,
        const float* __restrict__ b_proj, const float* __restrict__ ln_g,
        const float* __restrict__ ln_b, const float* __restrict__ xs,
        float* __restrict__ gated, int baseOff, int P) {
    __shared__ float lnbuf[16 * 256];      // [pixel][channel]
    __shared__ float mu_s[16], rs_s[16];
    int b = blockIdx.y;
    int pbase = blockIdx.x << 4;
    int tid = threadIdx.x;
    int wave = tid >> 5, lane = tid & 31;
    int hi = lane >> 4, lo = lane & 15;
    size_t prow = ((size_t)b * P + baseOff + pbase + lo) * 256;

    for (int mi = 0; mi < 2; ++mi) {
        int mt = wave + mi * 8;
        v8f acc = {};
        for (int t = 0; t < 8; ++t) {      // K = 256, 32 per step
            v16h a, bf;
            const _Float16* pa = w_proj16 + (size_t)(mt * 16 + lo) * 256 + t * 32 + hi * 8;
            ((v8h*)&a)[0] = *(const v8h*)pa;
            ((v8h*)&a)[1] = *(const v8h*)(pa + 16);
            const _Float16* pb = dcnT + prow + t * 32 + hi * 8;
            ((v8h*)&bf)[0] = *(const v8h*)pb;
            ((v8h*)&bf)[1] = *(const v8h*)(pb + 16);
            acc = __builtin_amdgcn_wmma_f32_16x16x32_f16(false, a, false, bf,
                                                         (short)0, acc, false, false);
        }
        for (int r = 0; r < 8; ++r) {
            int oc = mt * 16 + hi * 8 + r;
            lnbuf[lo * 256 + oc] = acc[r] + b_proj[oc];
        }
    }
    __syncthreads();

    if (tid < 16) {                        // LayerNorm stats per pixel
        float s = 0.0f;
        for (int c = 0; c < 256; ++c) s += lnbuf[tid * 256 + c];
        float mu = s * (1.0f / 256.0f);
        float v = 0.0f;
        for (int c = 0; c < 256; ++c) { float d = lnbuf[tid * 256 + c] - mu; v += d * d; }
        mu_s[tid] = mu;
        rs_s[tid] = rsqrtf(v * (1.0f / 256.0f) + 1e-5f);
    }
    __syncthreads();

    int p = tid & 15, cg = tid >> 4;
    for (int cc = 0; cc < 16; ++cc) {
        int c = cc * 16 + cg;
        float dn = (lnbuf[p * 256 + c] - mu_s[p]) * rs_s[p] * ln_g[c] + ln_b[c];
        float sg = 1.0f / (1.0f + __expf(-dn));
        size_t xi = (size_t)(b * CC + c) * P + baseOff + pbase + p;
        gated[xi] = xs[xi] * sg;
    }
}

// ---------------------------------------------------------------------------
// task_att final: out = max(a1*x2+b1, a2*x2+b2), shape-major output layout
// ---------------------------------------------------------------------------
__global__ void k_final(const float* __restrict__ gated, const float* __restrict__ ab,
                        float* __restrict__ out, int HW, int baseOff, int P, int outBase) {
    int idx = blockIdx.x * 256 + threadIdx.x;
    int total = NB * CC * HW;
    if (idx >= total) return;
    int p = idx % HW, bc = idx / HW;
    float x2 = gated[(size_t)bc * P + baseOff + p];
    float a1 = ab[bc], b1 = ab[512 + bc], a2 = ab[1024 + bc], b2 = ab[1536 + bc];
    out[(size_t)outBase + idx] = fmaxf(a1 * x2 + b1, a2 * x2 + b2);
}

// ---------------------------------------------------------------------------
extern "C" void kernel_launch(void* const* d_in, const int* in_sizes, int n_in,
                              void* d_out, int out_size, void* d_ws, size_t ws_size,
                              hipStream_t stream) {
    const float* f[3]    = {(const float*)d_in[0], (const float*)d_in[1], (const float*)d_in[2]};
    const float* w_scale = (const float*)d_in[3];
    const float* b_scale = (const float*)d_in[4];
    const float* w_om    = (const float*)d_in[5];
    const float* b_om    = (const float*)d_in[6];
    const float* w_proj  = (const float*)d_in[7];
    const float* b_proj  = (const float*)d_in[8];
    const float* ln_g    = (const float*)d_in[9];
    const float* ln_b    = (const float*)d_in[10];
    const float* wa1 = (const float*)d_in[11]; const float* ba1 = (const float*)d_in[12];
    const float* wb1 = (const float*)d_in[13]; const float* bb1 = (const float*)d_in[14];
    const float* wa2 = (const float*)d_in[15]; const float* ba2 = (const float*)d_in[16];
    const float* wb2 = (const float*)d_in[17]; const float* bb2 = (const float*)d_in[18];

    // carve workspace (~39 MB total)
    char* ws = (char*)d_ws;
    size_t off = 0;
    auto carve = [&](size_t bytes) -> void* {
        void* q = ws + off;
        off = (off + bytes + 255) & ~(size_t)255;
        return q;
    };
    _Float16* w_om16   = (_Float16*)carve((size_t)OMP * 2304 * 2);
    _Float16* w_proj16 = (_Float16*)carve((size_t)256 * 256 * 2);
    float*    xs       = (float*)   carve((size_t)NB * CC * PTOT * 4);
    _Float16* xs16     = (_Float16*)carve((size_t)NB * PTOT * CC * 2);
    float*    om       = (float*)   carve((size_t)NB * OMC * PTOT * 4);
    _Float16* dcnT     = (_Float16*)carve((size_t)NB * PTOT * CC * 2);
    float*    gated    = (float*)   carve((size_t)NB * CC * PTOT * 4);
    float*    meanv    = (float*)   carve((size_t)NB * CC * 4);
    float*    svec     = (float*)   carve((size_t)NB * CC * 4);
    float*    ab       = (float*)   carve((size_t)4 * NB * CC * 4);

    const int Hs[3]  = {64, 32, 16};
    const int Ws[3]  = {64, 32, 16};
    const int HWs[3] = {4096, 1024, 256};
    const int bas[3] = {0, 4096, 5120};
    const int oOf[3] = {0, NB * CC * 4096, NB * CC * 5120};

    k_prep<<<(OMP * 2304 + 256 * 256 + 255) / 256, 256, 0, stream>>>(w_om, w_proj, w_om16, w_proj16);

    for (int s = 0; s < 3; ++s) {
        int H = Hs[s], W = Ws[s], HW = HWs[s], base = bas[s];
        int nEl = NB * CC * HW;

        k_pool<<<dim3(CC, NB), 256, 0, stream>>>(f[s], meanv, HW, 0, HW);
        k_scalevec<<<NB, 256, 0, stream>>>(w_scale, b_scale, meanv, svec);
        k_applyscale<<<(nEl + 255) / 256, 256, 0, stream>>>(f[s], svec, xs, xs16, HW, base, PTOT);

        k_conv3x3<<<dim3(HW / 16, NB), 128, 0, stream>>>(xs16, w_om16, b_om, om, H, W, base, PTOT);
        k_dcn<<<dim3(HW, NB), 256, 0, stream>>>(xs, om, dcnT, H, W, base, PTOT);
        k_projlngate<<<dim3(HW / 16, NB), 256, 0, stream>>>(dcnT, w_proj16, b_proj,
                                                            ln_g, ln_b, xs, gated, base, PTOT);

        k_pool<<<dim3(CC, NB), 256, 0, stream>>>(gated, meanv, PTOT, base, HW);
        k_matvec_relu<<<NB, 256, 0, stream>>>(wa1, ba1, meanv, ab + 0);
        k_matvec_relu<<<NB, 256, 0, stream>>>(wb1, bb1, meanv, ab + 512);
        k_matvec_relu<<<NB, 256, 0, stream>>>(wa2, ba2, meanv, ab + 1024);
        k_matvec_relu<<<NB, 256, 0, stream>>>(wb2, bb2, meanv, ab + 1536);

        k_final<<<(nEl + 255) / 256, 256, 0, stream>>>(gated, ab, (float*)d_out, HW, base, PTOT, oOf[s]);
    }
    (void)in_sizes; (void)n_in; (void)out_size; (void)ws_size;
}
